// Attention_16655883174217
// MI455X (gfx1250) — compile-verified, matrix operand order checked by reading, not run on hardware
//
#include <hip/hip_runtime.h>
#include <hip/hip_bf16.h>

#define DIM     1024
#define NHEADS  16
#define HDIM    64
#define BATCH   4
#define SEQ     2048
#define MTOT    (BATCH*SEQ)   // 8192
#define NQKV    (3*DIM)       // 3072
#define ATTN_SCALE 0.125f     // 64^-0.5

typedef __attribute__((ext_vector_type(16))) __bf16 v16bf;
typedef __attribute__((ext_vector_type(8)))  __bf16 v8bf;
typedef __attribute__((ext_vector_type(8)))  float  v8f;

// ---------------------------------------------------------------------------
// WMMA helper: D = A(16x32 bf16) * B(32x16 bf16) + C(16x16 f32)
// ---------------------------------------------------------------------------
static __device__ __forceinline__ v8f wmma_bf16(v16bf a, v16bf b, v8f c) {
  return __builtin_amdgcn_wmma_f32_16x16x32_bf16(
      /*neg_a=*/false, a, /*neg_b=*/false, b,
      /*c_mod=*/(short)0, c, /*reuse_a=*/false, /*reuse_b=*/false);
}

// ---------------------------------------------------------------------------
// Fragment loader per CDNA5 ISA 7.12.2 (16-bit A-matrix 16x32):
//   lanes 0-15 : row = lane,    K = 0..7 (v0-3), 16..23 (v4-7)
//   lanes 16-31: row = lane-16, K = 8..15,       24..31
// Same layout serves B-frags by loading B^T row-major (row = N column).
// ---------------------------------------------------------------------------
static __device__ __forceinline__ v16bf frag_bf16(const __bf16* p, int ld) {
  const int lane = threadIdx.x & 31;
  const int r    = lane & 15;
  const int kofs = (lane >> 4) << 3;   // 0 or 8
  const __bf16* rp = p + (size_t)r * ld + kofs;
  v8bf lo = *(const v8bf*)(rp);        // K = kofs .. kofs+7
  v8bf hi = *(const v8bf*)(rp + 16);   // K = 16+kofs .. 16+kofs+7
  v16bf o;
#pragma unroll
  for (int i = 0; i < 8; ++i) { o[i] = lo[i]; o[i + 8] = hi[i]; }
  return o;
}

// ---------------------------------------------------------------------------
// Elementwise f32 -> bf16 cast (pre-pass for x)
// ---------------------------------------------------------------------------
__global__ void cast_bf16_kernel(const float* __restrict__ src,
                                 __bf16* __restrict__ dst, int n) {
  int idx = blockIdx.x * blockDim.x + threadIdx.x;
  if (idx < n) dst[idx] = (__bf16)src[idx];
}

// ---------------------------------------------------------------------------
// Weight transpose + f32->bf16:  wt[n][k] = (bf16) w[k][n]
// ---------------------------------------------------------------------------
__global__ void transpose_bf16_kernel(const float* __restrict__ w,
                                      __bf16* __restrict__ wt,
                                      int K, int N) {
  int idx = blockIdx.x * blockDim.x + threadIdx.x;
  if (idx >= K * N) return;
  int n = idx / K;
  int k = idx - n * K;
  wt[idx] = (__bf16)w[(size_t)k * N + n];
}

// ---------------------------------------------------------------------------
// QKV GEMM: [8192 x 1024]bf16 x [1024 x 3072] + bias, scatter Q/K/V^T (bf16).
// WG = 256 threads = 8 waves (2 row-waves x 4 col-waves).
// Wave tile = 64x64 (4x4 WMMA tiles). WG tile = 128x256.
// ---------------------------------------------------------------------------
__global__ __launch_bounds__(256) void qkv_gemm_kernel(
    const __bf16* __restrict__ xb,    // [MTOT, DIM] bf16
    const __bf16* __restrict__ wt,    // [NQKV, DIM]  (w_qkv^T bf16)
    const float* __restrict__ bias,   // [NQKV]
    __bf16* __restrict__ Qb,          // [B,H,T,D] (pre-scaled)
    __bf16* __restrict__ Kb,          // [B,H,T,D]
    __bf16* __restrict__ Vt)          // [B,H,D,T]
{
  const int lane = threadIdx.x & 31;
  const int wave = threadIdx.x >> 5;
  const int m0 = blockIdx.x * 128 + (wave >> 2) * 64;
  const int n0 = blockIdx.y * 256 + (wave & 3) * 64;

  v8f acc[4][4];
#pragma unroll
  for (int i = 0; i < 4; ++i)
#pragma unroll
    for (int j = 0; j < 4; ++j) acc[i][j] = (v8f)0.0f;

  for (int k0 = 0; k0 < DIM; k0 += 32) {
    v16bf a[4], b[4];
#pragma unroll
    for (int i = 0; i < 4; ++i)
      a[i] = frag_bf16(xb + (size_t)(m0 + i * 16) * DIM + k0, DIM);
#pragma unroll
    for (int j = 0; j < 4; ++j)
      b[j] = frag_bf16(wt + (size_t)(n0 + j * 16) * DIM + k0, DIM);
#pragma unroll
    for (int i = 0; i < 4; ++i)
#pragma unroll
      for (int j = 0; j < 4; ++j)
        acc[i][j] = wmma_bf16(a[i], b[j], acc[i][j]);
  }

  // Epilogue. n0 is 64-aligned => this wave's 64 cols live in exactly one
  // (section s, head hh): n = s*1024 + hh*64 + d.
  const int s     = n0 / DIM;              // 0=Q, 1=K, 2=V
  const int hh    = (n0 % DIM) / HDIM;
  const int rbase = (lane >> 4) * 8;       // C-layout: vgpr e -> row rbase+e
  const int cl    = lane & 15;
#pragma unroll
  for (int j = 0; j < 4; ++j) {
    const int n  = n0 + j * 16 + cl;
    const int d  = j * 16 + cl;            // n0 % 64 == 0
    const float bv = bias[n];
#pragma unroll
    for (int i = 0; i < 4; ++i) {
#pragma unroll
      for (int e = 0; e < 8; ++e) {
        const int m  = m0 + i * 16 + rbase + e;
        const int bb = m >> 11;            // m / SEQ
        const int t  = m & (SEQ - 1);
        const float v = acc[i][j][e] + bv;
        if (s == 0) {
          Qb[((size_t)(bb * NHEADS + hh) * SEQ + t) * HDIM + d] =
              (__bf16)(v * ATTN_SCALE);
        } else if (s == 1) {
          Kb[((size_t)(bb * NHEADS + hh) * SEQ + t) * HDIM + d] = (__bf16)v;
        } else {
          Vt[((size_t)(bb * NHEADS + hh) * HDIM + d) * SEQ + t] = (__bf16)v;
        }
      }
    }
  }
}

// ---------------------------------------------------------------------------
// Flash attention. Grid: (SEQ/128, BATCH*NHEADS). 8 waves/WG, each wave owns
// a 16-row q-block and streams 32-key blocks with online softmax.
// ---------------------------------------------------------------------------
__global__ __launch_bounds__(256) void attn_kernel(
    const __bf16* __restrict__ Qb,   // [B,H,T,D] (pre-scaled by 1/8)
    const __bf16* __restrict__ Kb,   // [B,H,T,D]
    const __bf16* __restrict__ Vt,   // [B,H,D,T]
    __bf16* __restrict__ Z)          // [B,T,C]
{
  __shared__ __bf16 lds_p[8][16][32];   // per-wave P tile (C-layout -> A-layout)

  const int lane = threadIdx.x & 31;
  const int wave = threadIdx.x >> 5;
  const int bh   = blockIdx.y;                       // b*NHEADS + h
  const int q0   = blockIdx.x * 128 + wave * 16;

  const __bf16* Qh = Qb + (size_t)bh * SEQ * HDIM;
  const __bf16* Kh = Kb + (size_t)bh * SEQ * HDIM;
  const __bf16* Vh = Vt + (size_t)bh * HDIM * SEQ;

  v16bf qf[2];
  qf[0] = frag_bf16(Qh + (size_t)q0 * HDIM +  0, HDIM);  // d 0..31
  qf[1] = frag_bf16(Qh + (size_t)q0 * HDIM + 32, HDIM);  // d 32..63

  v8f o[4];
#pragma unroll
  for (int t = 0; t < 4; ++t) o[t] = (v8f)0.0f;
  float m_[8], l_[8];
#pragma unroll
  for (int j = 0; j < 8; ++j) { m_[j] = -1e30f; l_[j] = 0.0f; }

  const int rbase = (lane >> 4) * 8;
  const int cl    = lane & 15;

  for (int k0 = 0; k0 < SEQ; k0 += 32) {
    // S = Q * K^T  (two 16x16 score tiles, K-dim = HDIM = 2 WMMAs each)
    v8f s0 = (v8f)0.0f, s1 = (v8f)0.0f;
    s0 = wmma_bf16(qf[0], frag_bf16(Kh + (size_t)(k0)      * HDIM +  0, HDIM), s0);
    s0 = wmma_bf16(qf[1], frag_bf16(Kh + (size_t)(k0)      * HDIM + 32, HDIM), s0);
    s1 = wmma_bf16(qf[0], frag_bf16(Kh + (size_t)(k0 + 16) * HDIM +  0, HDIM), s1);
    s1 = wmma_bf16(qf[1], frag_bf16(Kh + (size_t)(k0 + 16) * HDIM + 32, HDIM), s1);

    // Online softmax. Row rbase+j is spread over the 16 lanes of this group
    // within vgpr slot j -> reduce with shfl_xor masks 1,2,4,8.
    float alpha[8];
#pragma unroll
    for (int j = 0; j < 8; ++j) {
      float mx = fmaxf(s0[j], s1[j]);
      mx = fmaxf(mx, __shfl_xor(mx, 1, 32));
      mx = fmaxf(mx, __shfl_xor(mx, 2, 32));
      mx = fmaxf(mx, __shfl_xor(mx, 4, 32));
      mx = fmaxf(mx, __shfl_xor(mx, 8, 32));
      const float mn = fmaxf(m_[j], mx);
      const float a  = __expf(m_[j] - mn);
      const float p0 = __expf(s0[j] - mn);
      const float p1 = __expf(s1[j] - mn);
      float rs = p0 + p1;
      rs += __shfl_xor(rs, 1, 32);
      rs += __shfl_xor(rs, 2, 32);
      rs += __shfl_xor(rs, 4, 32);
      rs += __shfl_xor(rs, 8, 32);
      l_[j] = l_[j] * a + rs;
      m_[j] = mn;
      alpha[j] = a;
      lds_p[wave][rbase + j][cl]      = (__bf16)p0;
      lds_p[wave][rbase + j][cl + 16] = (__bf16)p1;
    }

    // Rescale running output, then O += P * V (4 d-tiles, K = 32 keys).
#pragma unroll
    for (int t = 0; t < 4; ++t)
#pragma unroll
      for (int e = 0; e < 8; ++e) o[t][e] *= alpha[e];

    const v16bf pf = frag_bf16(&lds_p[wave][0][0], 32);
#pragma unroll
    for (int t = 0; t < 4; ++t) {
      const v16bf vf = frag_bf16(Vh + (size_t)(t * 16) * SEQ + k0, SEQ);
      o[t] = wmma_bf16(pf, vf, o[t]);
    }
  }

  // Normalize and write Z[b][t][h*64 + d] as bf16.
  const int bb = bh >> 4;           // / NHEADS
  const int hh = bh & (NHEADS - 1);
  float linv[8];
#pragma unroll
  for (int e = 0; e < 8; ++e) linv[e] = 1.0f / l_[e];
#pragma unroll
  for (int t = 0; t < 4; ++t) {
#pragma unroll
    for (int e = 0; e < 8; ++e) {
      const int row = q0 + rbase + e;
      const int d   = t * 16 + cl;
      const float val = o[t][e] * linv[e];
      Z[((size_t)bb * SEQ + row) * DIM + hh * HDIM + d] = (__bf16)val;
    }
  }
}

// ---------------------------------------------------------------------------
// Output projection: out = Z @ w_proj + b_proj   (f32 output)
// Wave tile 64x64 (4x4), WG tile 128x256.
// ---------------------------------------------------------------------------
__global__ __launch_bounds__(256) void proj_gemm_kernel(
    const __bf16* __restrict__ Z,     // [MTOT, DIM]
    const __bf16* __restrict__ wt,    // [DIM, DIM] (w_proj^T bf16)
    const float* __restrict__ bias,   // [DIM]
    float* __restrict__ out)          // [MTOT, DIM]
{
  const int lane = threadIdx.x & 31;
  const int wave = threadIdx.x >> 5;
  const int m0 = blockIdx.x * 128 + (wave >> 2) * 64;
  const int n0 = blockIdx.y * 256 + (wave & 3) * 64;

  v8f acc[4][4];
#pragma unroll
  for (int i = 0; i < 4; ++i)
#pragma unroll
    for (int j = 0; j < 4; ++j) acc[i][j] = (v8f)0.0f;

  for (int k0 = 0; k0 < DIM; k0 += 32) {
    v16bf a[4], b[4];
#pragma unroll
    for (int i = 0; i < 4; ++i)
      a[i] = frag_bf16(Z + (size_t)(m0 + i * 16) * DIM + k0, DIM);
#pragma unroll
    for (int j = 0; j < 4; ++j)
      b[j] = frag_bf16(wt + (size_t)(n0 + j * 16) * DIM + k0, DIM);
#pragma unroll
    for (int i = 0; i < 4; ++i)
#pragma unroll
      for (int j = 0; j < 4; ++j)
        acc[i][j] = wmma_bf16(a[i], b[j], acc[i][j]);
  }

  const int rbase = (lane >> 4) * 8;
  const int cl    = lane & 15;
#pragma unroll
  for (int j = 0; j < 4; ++j) {
    const int n = n0 + j * 16 + cl;
    const float bv = bias[n];
#pragma unroll
    for (int i = 0; i < 4; ++i)
#pragma unroll
      for (int e = 0; e < 8; ++e) {
        const int m = m0 + i * 16 + rbase + e;
        out[(size_t)m * DIM + n] = acc[i][j][e] + bv;
      }
  }
}

// ---------------------------------------------------------------------------
extern "C" void kernel_launch(void* const* d_in, const int* in_sizes, int n_in,
                              void* d_out, int out_size, void* d_ws, size_t ws_size,
                              hipStream_t stream) {
  (void)in_sizes; (void)n_in; (void)out_size; (void)ws_size;
  const float* x      = (const float*)d_in[0];
  const float* w_qkv  = (const float*)d_in[1];
  const float* b_qkv  = (const float*)d_in[2];
  const float* w_proj = (const float*)d_in[3];
  const float* b_proj = (const float*)d_in[4];
  float* out = (float*)d_out;

  char* ws = (char*)d_ws;
  __bf16* wqkvt  = (__bf16*)ws; ws += (size_t)NQKV * DIM * sizeof(__bf16);
  __bf16* wprojt = (__bf16*)ws; ws += (size_t)DIM  * DIM * sizeof(__bf16);
  __bf16* Xb     = (__bf16*)ws; ws += (size_t)MTOT * DIM * sizeof(__bf16);
  __bf16* Qb     = (__bf16*)ws; ws += (size_t)BATCH * NHEADS * SEQ * HDIM * sizeof(__bf16);
  __bf16* Kb     = (__bf16*)ws; ws += (size_t)BATCH * NHEADS * SEQ * HDIM * sizeof(__bf16);
  __bf16* Vt     = (__bf16*)ws; ws += (size_t)BATCH * NHEADS * HDIM * SEQ * sizeof(__bf16);
  __bf16* Zb     = (__bf16*)ws; ws += (size_t)MTOT * DIM * sizeof(__bf16);

  // 1) input cast + weight transposes (f32 -> bf16, N-major for B-frags)
  {
    int n = MTOT * DIM;
    cast_bf16_kernel<<<(n + 255) / 256, 256, 0, stream>>>(x, Xb, n);
  }
  {
    int n = NQKV * DIM;
    transpose_bf16_kernel<<<(n + 255) / 256, 256, 0, stream>>>(w_qkv, wqkvt, DIM, NQKV);
  }
  {
    int n = DIM * DIM;
    transpose_bf16_kernel<<<(n + 255) / 256, 256, 0, stream>>>(w_proj, wprojt, DIM, DIM);
  }

  // 2) QKV projection, scatter into per-head Q / K / V^T
  qkv_gemm_kernel<<<dim3(MTOT / 128, NQKV / 256), 256, 0, stream>>>(
      Xb, wqkvt, b_qkv, Qb, Kb, Vt);

  // 3) flash attention
  attn_kernel<<<dim3(SEQ / 128, BATCH * NHEADS), 256, 0, stream>>>(Qb, Kb, Vt, Zb);

  // 4) output projection
  proj_gemm_kernel<<<dim3(MTOT / 128, DIM / 256), 256, 0, stream>>>(
      Zb, wprojt, b_proj, out);
}
